// GlimpseNetwork_24730421690989
// MI455X (gfx1250) — matrix operand order repaired and checked by reading, not compiled, not run on hardware
//
#include <hip/hip_runtime.h>

// ---------------------------------------------------------------------------
// CDNA5 / gfx1250 Glimpse Network. wave32, WMMA bf16 (v_wmma_f32_16x16x32_bf16)
// GEMMs: 32x32 output tile per wave (2x2 WMMA sub-tiles, 4 f32 accumulators)
// ---------------------------------------------------------------------------

typedef __attribute__((ext_vector_type(16))) __bf16 v16bf;
typedef __attribute__((ext_vector_type(8)))  float  v8f;

struct alignas(16) U4 { unsigned int x, y, z, w; };
union Vec32B { v16bf v; U4 u[2]; };

static __device__ __forceinline__ unsigned short f32_to_bf16(float f) {
    unsigned int u = __float_as_uint(f);
    u += 0x7fffu + ((u >> 16) & 1u);   // round-to-nearest-even
    return (unsigned short)(u >> 16);
}

// ---------------------------------------------------------------------------
// Foveation: gather 3 zoom crops, bilinear-downsample, emit bf16 g (256x9216).
// zoom0: identity; zoom1: mean of 2x2 at (2y,2x); zoom2: mean of 2x2 at (4y+1,4x+1)
// ---------------------------------------------------------------------------
__global__ void foveate_kernel(const float* __restrict__ img,
                               const float* __restrict__ loc,
                               unsigned short* __restrict__ g) {
    const int GD = 9216;
    int idx = blockIdx.x * blockDim.x + threadIdx.x;
    if (idx >= 256 * GD) return;
    int b   = idx / GD;
    int r   = idx - b * GD;
    int z   = r / 3072;
    int rem = r - z * 3072;
    int c   = rem / 1024;
    int yx  = rem - c * 1024;
    int y = yx >> 5, x = yx & 31;

    float dr = (loc[2 * b]     + 1.0f) * 112.0f;   // (loc+1)*0.5*224
    float dc = (loc[2 * b + 1] + 1.0f) * 112.0f;
    const float* im = img + ((size_t)b * 3 + c) * 224 * 224;

    float val;
    if (z == 0) {
        int top = (int)(dr - 16.0f), left = (int)(dc - 16.0f);
        val = im[(top + y) * 224 + (left + x)];
    } else if (z == 1) {
        int top = (int)(dr - 32.0f), left = (int)(dc - 32.0f);
        int rr = top + 2 * y, cc = left + 2 * x;
        val = 0.25f * (im[rr * 224 + cc]       + im[rr * 224 + cc + 1] +
                       im[(rr + 1) * 224 + cc] + im[(rr + 1) * 224 + cc + 1]);
    } else {
        int top = (int)(dr - 64.0f), left = (int)(dc - 64.0f);
        int rr = top + 4 * y + 1, cc = left + 4 * x + 1;
        val = 0.25f * (im[rr * 224 + cc]       + im[rr * 224 + cc + 1] +
                       im[(rr + 1) * 224 + cc] + im[(rr + 1) * 224 + cc + 1]);
    }
    g[idx] = f32_to_bf16(val);
}

// ---------------------------------------------------------------------------
// Transpose fp32 (K x N, row-major) -> bf16 (N x K), LDS-tiled for coalescing.
// ---------------------------------------------------------------------------
__global__ void transpose_to_bf16(const float* __restrict__ src,
                                  unsigned short* __restrict__ dst,
                                  int K, int N) {
    __shared__ float tile[32][33];
    int k0 = blockIdx.y * 32, n0 = blockIdx.x * 32;
    int tx = threadIdx.x, ty = threadIdx.y;
    int k = k0 + ty, n = n0 + tx;
    if (k < K && n < N) tile[ty][tx] = src[(size_t)k * N + n];
    __syncthreads();
    int on = n0 + ty, ok = k0 + tx;
    if (on < N && ok < K) dst[(size_t)on * K + ok] = f32_to_bf16(tile[tx][ty]);
}

// ---------------------------------------------------------------------------
// hl1 = relu(loc @ W2 + b2), K=2 -> plain VALU, bf16 output (256 x 256).
// ---------------------------------------------------------------------------
__global__ void loc_fc_kernel(const float* __restrict__ loc,
                              const float* __restrict__ W2,
                              const float* __restrict__ b2,
                              unsigned short* __restrict__ hl1) {
    int idx = blockIdx.x * blockDim.x + threadIdx.x;
    if (idx >= 256 * 256) return;
    int b = idx >> 8, j = idx & 255;
    float v = loc[2 * b] * W2[j] + loc[2 * b + 1] * W2[256 + j] + b2[j];
    hl1[idx] = f32_to_bf16(fmaxf(v, 0.0f));
}

// ---------------------------------------------------------------------------
// Operand fetch helpers for v_wmma_f32_16x16x32_bf16 VGPR layouts (wave32):
//  A (16x32): lane<16 -> row l16, K {k0..k0+7, k0+16..k0+23}; lane>=16 -> +8
//  B (32x16): lane<16 -> col l16, K {k0..k0+7, k0+16..k0+23}; lane>=16 -> +8
// ---------------------------------------------------------------------------
static __device__ __forceinline__ v16bf load_a(const unsigned short* row, int k0, int hi) {
    Vec32B a;
    a.u[0] = *(const U4*)(row + k0 + 8 * hi);
    a.u[1] = *(const U4*)(row + k0 + 16 + 8 * hi);
    return a.v;
}
static __device__ __forceinline__ v16bf load_b(const unsigned short* row, int k0, int hi) {
    Vec32B b;
    b.u[0] = *(const U4*)(row + k0 + 8 * hi);
    b.u[1] = *(const U4*)(row + k0 + 16 + 8 * hi);
    return b.v;
}

// ---------------------------------------------------------------------------
// WMMA bf16 GEMM: out[M,N] = epilogue(A[M,K] @ Bt[N,K]^T + bias)
// One wave per 32x32 tile (2x2 sub-tiles of 16x16). A row-major bf16,
// Bt = transposed weights (N x K) bf16. M, N multiples of 32; K multiple of 32.
// F32OUT=false : out = bf16( relu(acc + bias) )
// F32OUT=true  : out = f32 ( relu(acc + bias) ), and if addend != null:
//                out = relu( relu(acc+bias) + addend )   (final fusion)
// ---------------------------------------------------------------------------
template <bool F32OUT>
__global__ void gemm_wmma_bf16(const unsigned short* __restrict__ A,
                               const unsigned short* __restrict__ Bt,
                               const float* __restrict__ bias,
                               const float* __restrict__ addend,
                               void* __restrict__ out,
                               int M, int N, int K) {
    const int n0   = blockIdx.x * 32;
    const int m0   = blockIdx.y * 32;
    const int lane = threadIdx.x;      // 0..31 (wave32)
    const int hi   = lane >> 4;        // lane half: 0 or 1
    const int l16  = lane & 15;

    const unsigned short* arow0 = A  + (size_t)(m0 + l16) * K;
    const unsigned short* arow1 = A  + (size_t)(m0 + 16 + l16) * K;
    const unsigned short* brow0 = Bt + (size_t)(n0 + l16) * K;
    const unsigned short* brow1 = Bt + (size_t)(n0 + 16 + l16) * K;

    v8f acc00 = {0.f, 0.f, 0.f, 0.f, 0.f, 0.f, 0.f, 0.f};
    v8f acc01 = acc00, acc10 = acc00, acc11 = acc00;

    for (int k0 = 0; k0 < K; k0 += 32) {
        v16bf a0 = load_a(arow0, k0, hi);
        v16bf a1 = load_a(arow1, k0, hi);
        v16bf b0 = load_b(brow0, k0, hi);
        v16bf b1 = load_b(brow1, k0, hi);
        acc00 = __builtin_amdgcn_wmma_f32_16x16x32_bf16(false, a0, false, b0, (short)0, acc00, false, false);
        acc01 = __builtin_amdgcn_wmma_f32_16x16x32_bf16(false, a0, false, b1, (short)0, acc01, false, false);
        acc10 = __builtin_amdgcn_wmma_f32_16x16x32_bf16(false, a1, false, b0, (short)0, acc10, false, false);
        acc11 = __builtin_amdgcn_wmma_f32_16x16x32_bf16(false, a1, false, b1, (short)0, acc11, false, false);
    }

    // C/D layout: element r -> row (tile_m + r + 8*hi), col (tile_n + l16)
    v8f accs[2][2] = {{acc00, acc01}, {acc10, acc11}};
#pragma unroll
    for (int mt = 0; mt < 2; ++mt) {
#pragma unroll
        for (int nt = 0; nt < 2; ++nt) {
            const int col = n0 + 16 * nt + l16;
            const float bv = bias ? bias[col] : 0.0f;
            v8f acc = accs[mt][nt];
#pragma unroll
            for (int r = 0; r < 8; ++r) {
                int row = m0 + 16 * mt + r + 8 * hi;
                float v = fmaxf(acc[r] + bv, 0.0f);
                if (F32OUT) {
                    if (addend) v = fmaxf(v + addend[(size_t)row * N + col], 0.0f);
                    ((float*)out)[(size_t)row * N + col] = v;
                } else {
                    ((unsigned short*)out)[(size_t)row * N + col] = f32_to_bf16(v);
                }
            }
        }
    }
}

// ---------------------------------------------------------------------------
// Workspace layout (bytes, all 256-aligned)
// ---------------------------------------------------------------------------
#define G_OFF    ((size_t)0)            // g bf16        256*9216*2  = 4718592
#define W1T_OFF  ((size_t)4718592)      // W1T bf16      9216*1024*2 = 18874368
#define W3T_OFF  ((size_t)23592960)     // W3T bf16      1024*1280*2 = 2621440
#define W4T_OFF  ((size_t)26214400)     // W4T bf16      256*1280*2  = 655360
#define HG1_OFF  ((size_t)26869760)     // hg1 bf16      256*1024*2  = 524288
#define HL1_OFF  ((size_t)27394048)     // hl1 bf16      256*256*2   = 131072
#define HL2_OFF  ((size_t)27525120)     // hl2 f32       256*1280*4  = 1310720

extern "C" void kernel_launch(void* const* d_in, const int* in_sizes, int n_in,
                              void* d_out, int out_size, void* d_ws, size_t ws_size,
                              hipStream_t stream) {
    const float* images    = (const float*)d_in[0];
    const float* locations = (const float*)d_in[1];
    const float* W1 = (const float*)d_in[2];
    const float* b1 = (const float*)d_in[3];
    const float* W2 = (const float*)d_in[4];
    const float* b2 = (const float*)d_in[5];
    const float* W3 = (const float*)d_in[6];
    const float* b3 = (const float*)d_in[7];
    const float* W4 = (const float*)d_in[8];
    const float* b4 = (const float*)d_in[9];

    char* ws = (char*)d_ws;
    unsigned short* g   = (unsigned short*)(ws + G_OFF);
    unsigned short* W1T = (unsigned short*)(ws + W1T_OFF);
    unsigned short* W3T = (unsigned short*)(ws + W3T_OFF);
    unsigned short* W4T = (unsigned short*)(ws + W4T_OFF);
    unsigned short* hg1 = (unsigned short*)(ws + HG1_OFF);
    unsigned short* hl1 = (unsigned short*)(ws + HL1_OFF);
    float*          hl2 = (float*)(ws + HL2_OFF);
    float*          out = (float*)d_out;

    // 1) Foveation -> g bf16 (256 x 9216)
    foveate_kernel<<<(256 * 9216 + 255) / 256, 256, 0, stream>>>(images, locations, g);

    // 2) Weight transpose + bf16 convert (L2-resident afterwards)
    transpose_to_bf16<<<dim3(1024 / 32, 9216 / 32), dim3(32, 32), 0, stream>>>(W1, W1T, 9216, 1024);
    transpose_to_bf16<<<dim3(1280 / 32, 1024 / 32), dim3(32, 32), 0, stream>>>(W3, W3T, 1024, 1280);
    transpose_to_bf16<<<dim3(1280 / 32,  256 / 32), dim3(32, 32), 0, stream>>>(W4, W4T,  256, 1280);

    // 3) hl1 = relu(loc @ W2 + b2) -> bf16 (256 x 256)
    loc_fc_kernel<<<(256 * 256 + 255) / 256, 256, 0, stream>>>(locations, W2, b2, hl1);

    // 4) hg1 = relu(g @ W1 + b1) -> bf16 (256 x 1024)    [WMMA, K=9216]
    gemm_wmma_bf16<false><<<dim3(1024 / 32, 256 / 32), 32, 0, stream>>>(
        g, W1T, b1, nullptr, (void*)hg1, 256, 1024, 9216);

    // 5) hl2 = relu(hl1 @ W4 + b4) -> f32 (256 x 1280)   [WMMA, K=256]
    gemm_wmma_bf16<true><<<dim3(1280 / 32, 256 / 32), 32, 0, stream>>>(
        hl1, W4T, b4, nullptr, (void*)hl2, 256, 1280, 256);

    // 6) out = relu( relu(hg1 @ W3 + b3) + hl2 ) -> f32 (256 x 1280)  [WMMA, fused]
    gemm_wmma_bf16<true><<<dim3(1280 / 32, 256 / 32), 32, 0, stream>>>(
        hg1, W3T, b3, hl2, (void*)out, 256, 1280, 1024);
}
